// VoxelBackBone8x_23244363006084
// MI455X (gfx1250) — matrix operand order repaired and checked by reading, not compiled, not run on hardware
//
#include <hip/hip_runtime.h>

// ---------------------------------------------------------------------------
// Types for CDNA5 WMMA (wave32): A/B = 16 halves (8 VGPRs), C/D = 8 floats.
// ---------------------------------------------------------------------------
typedef __attribute__((ext_vector_type(16))) _Float16 v16h;
typedef __attribute__((ext_vector_type(8)))  float    v8f;
typedef __attribute__((ext_vector_type(4)))  unsigned uv4;

// ---------------------------------------------------------------------------
// Small prep kernels
// ---------------------------------------------------------------------------

// Re-tile weights [t][cinR][cout] f32 -> fragment-ready f16 layout with both
// cin padding (cinR -> cinP) and K padding (to NCHUNK*32):
//   dst[((nt*NCHUNK + ch)*16 + n)*32 + kk],  k = ch*32+kk = t*cinP + c
// A lane's B fragment (N = n, K = 8*half..+7 and 16+8*half..+7) is two 16B runs in memory.
__global__ void wtrans_kernel(const float* __restrict__ w, _Float16* __restrict__ o,
                              int NTAPS, int CINR, int CINP, int COUT, int NCHUNK) {
    int i = blockIdx.x * blockDim.x + threadIdx.x;
    int total = NCHUNK * 32 * COUT;
    if (i >= total) return;
    int kk = i & 31;
    int n  = (i >> 5) & 15;
    int cn = i >> 9;              // nt*NCHUNK + ch
    int ch = cn % NCHUNK;
    int nt = cn / NCHUNK;
    int k = ch * 32 + kk;         // padded K index
    int t = k / CINP;
    int c = k - t * CINP;
    int cout = nt * 16 + n;
    float v = 0.f;
    if (t < NTAPS && c < CINR) v = w[((size_t)t * CINR + c) * COUT + cout];
    o[i] = (_Float16)v;
}

__global__ void bnprep_kernel(const float* __restrict__ bn, float* __restrict__ scale,
                              float* __restrict__ shift, int C) {
    int c = blockIdx.x * blockDim.x + threadIdx.x;
    if (c < C) {
        float g = bn[c], b = bn[C + c], m = bn[2 * C + c], v = bn[3 * C + c];
        float s = g * rsqrtf(v + 1e-4f);
        scale[c] = s;
        shift[c] = b - m * s;
    }
}

// zero-fill in 32-bit words (all our f16 regions have even element counts)
__global__ void fill_zero_w_kernel(unsigned* __restrict__ p, long long nw) {
    long long i = (long long)blockIdx.x * blockDim.x + threadIdx.x;
    if (i < nw) p[i] = 0u;
}

// Scatter voxels into the haloed dense grid (16 channels: 4 real + 12 zero pad)
__global__ void scatter_kernel(const float* __restrict__ feats, const int* __restrict__ coords,
                               int n, _Float16* __restrict__ dense, _Float16* __restrict__ mask,
                               int H, int W) {
    int i = blockIdx.x * blockDim.x + threadIdx.x;
    if (i >= n) return;
    int z = coords[i * 4 + 1], y = coords[i * 4 + 2], x = coords[i * 4 + 3];
    size_t cellH = ((size_t)(z + 1) * (H + 2) + (y + 1)) * (W + 2) + (x + 1);
    size_t cell  = ((size_t)z * H + y) * W + x;
    #pragma unroll
    for (int c = 0; c < 4; ++c) dense[cellH * 16 + c] = (_Float16)feats[i * 4 + c];
    mask[cell] = (_Float16)1.f;
}

__global__ void maskpool_kernel(const _Float16* __restrict__ in, int IH, int IW,
                                _Float16* __restrict__ out, int OD, int OH, int OW,
                                int KD, int KH, int KW, int S) {
    int i = blockIdx.x * blockDim.x + threadIdx.x;
    int Mtot = OD * OH * OW;
    if (i >= Mtot) return;
    int od = i / (OH * OW);
    int rem = i - od * (OH * OW);
    int oh = rem / OW, ow = rem - (rem / OW) * OW;
    float mx = 0.f;
    for (int kd = 0; kd < KD; ++kd)
        for (int kh = 0; kh < KH; ++kh)
            for (int kw = 0; kw < KW; ++kw) {
                size_t off = (((size_t)(od * S + kd) * IH + (oh * S + kh)) * IW + (ow * S + kw));
                float v = (float)in[off];
                mx = v > mx ? v : mx;
            }
    out[i] = (_Float16)mx;
}

// ---------------------------------------------------------------------------
// Implicit-GEMM 3D conv, fused BN+ReLU+mask, via v_wmma_f32_16x16x32_f16.
//   Activations f16 with a 1-voxel zero halo (no bounds checks, EXEC all-1s).
//   M = output voxels (16/tile, 4 tiles per block, one per wave)
//   N = cout tile of 16 (blockIdx.y)
//   K-chunk loop fully unrolled: tap decode const-folds, tap bases are
//   uniform (SALU), lane offsets loop-invariant -> pure b128 load + wmma loop.
//   B fragments load directly from the fragment-ready weight layout: no LDS,
//   no barriers. CIN >= 16 always (layer 1a is channel-padded to 16).
// ---------------------------------------------------------------------------
template <int CIN, int COUT, int KD, int KH, int KW, int S,
          int PD, int PH, int PW, bool OUT_F32>
__global__ void __launch_bounds__(128)
conv3d_wmma(const _Float16* __restrict__ in, int ID, int IH, int IW,
            void* __restrict__ outp, int OD, int OH, int OW,
            const _Float16* __restrict__ wgt,        // [nt][NCHUNK][16][32] f16
            const float* __restrict__ scale, const float* __restrict__ shift,
            const _Float16* __restrict__ maskO) {
    static_assert(CIN >= 16 && (CIN == 16 || CIN % 32 == 0), "CIN must be 16 or mult of 32");
    constexpr int NTAPS = KD * KH * KW;
    constexpr int KHW   = KH * KW;
    constexpr int KTOT  = NTAPS * CIN;
    constexpr int NCHUNK = (KTOT + 31) / 32;

    const int tid  = threadIdx.x;
    const int wave = tid >> 5;
    const int lane = tid & 31;
    const int half = lane >> 4;     // lane half: K sub-range (A/B), +8 row offset (C/D)
    const int l16  = lane & 15;     // A: M row; B/C/D: N column
    const int Mtot = OD * OH * OW;
    const int tileBase = (blockIdx.x * 4 + wave) * 16;

    // This lane's A-matrix row -> output voxel -> input origin (haloed frame).
    const int IHH = IH + 2, IWH = IW + 2;
    int m = tileBase + l16;
    int mm = (m < Mtot) ? m : (Mtot - 1);     // clamp: keeps all addresses in-bounds
    int od = mm / (OH * OW);
    int rem = mm - od * (OH * OW);
    int oh = rem / OW;
    int ow = rem - oh * OW;
    const int orgVox = (od * S - PD + 1) * IHH * IWH + (oh * S - PH + 1) * IWH
                       + (ow * S - PW + 1);

    // Loop-invariant per-lane offset (elements). 32-bit: grids < 2^26 elems.
    const int laneOff = orgVox * CIN + 8 * half;

    // Uniform weight tile base for this N-tile; divergent per-lane fragment offset.
    const _Float16* wTile = wgt + (size_t)blockIdx.y * (NCHUNK * 512);
    const int wLane = l16 * 32 + 8 * half;

    __builtin_prefetch(wTile + wLane, 0, 3);

    v8f acc = {0.f, 0.f, 0.f, 0.f, 0.f, 0.f, 0.f, 0.f};

    #pragma unroll
    for (int ci = 0; ci < NCHUNK; ++ci) {
        // ---- B fragment: two global_load_b128 from fragment-ready layout ----
        union { v16h h; uv4 q[2]; } bf;
        bf.q[0] = *(const uv4*)(wTile + wLane + ci * 512);
        bf.q[1] = *(const uv4*)(wTile + wLane + ci * 512 + 16);

        // ---- A fragment: compile-time tap decode, uniform tap base + laneOff ----
        union { v16h h; uv4 q[2]; } af;
        if constexpr (CIN >= 32) {
            // whole chunk inside one tap (CIN % 32 == 0)
            const int t  = (ci * 32) / CIN;
            const int c0 = (ci * 32) % CIN;
            const int kd = t / KHW, tr = t % KHW, kh = tr / KW, kw = tr % KW;
            const _Float16* tp = in + (((size_t)kd * IHH + kh) * IWH + kw) * CIN + c0;
            af.q[0] = *(const uv4*)(tp + laneOff);          // K = 8h .. 8h+7
            af.q[1] = *(const uv4*)(tp + laneOff + 16);     // K = 16+8h ..
        } else {
            // CIN == 16: chunk covers taps 2ci, 2ci+1; seg s -> tap 2ci+s, c = 8h..
            #pragma unroll
            for (int s = 0; s < 2; ++s) {
                int t = 2 * ci + s;
                t = (t < NTAPS) ? t : (NTAPS - 1);          // pad rows: B is zero there
                const int kd = t / KHW, tr = t % KHW, kh = tr / KW, kw = tr % KW;
                const _Float16* tp = in + (((size_t)kd * IHH + kh) * IWH + kw) * 16;
                af.q[s] = *(const uv4*)(tp + laneOff);
            }
        }

        // ---- D = A x B + C : v_wmma_f32_16x16x32_f16 ----
        acc = __builtin_amdgcn_wmma_f32_16x16x32_f16(
            /*neg_a=*/false, af.h, /*neg_b=*/false, bf.h,
            /*c_mod=*/(short)0, acc, /*reuse_a=*/false, /*reuse_b=*/false);
    }

    // ---- epilogue: BN (scale/shift) + ReLU + mask, per C/D layout ----
    const int cout = blockIdx.y * 16 + l16;
    const float sc = scale[cout];
    const float sh = shift[cout];
    #pragma unroll
    for (int r = 0; r < 8; ++r) {
        int ms = tileBase + r + 8 * half;   // D row M = r + 8*half
        if (ms < Mtot) {
            float y = acc[r] * sc + sh;
            y = y > 0.f ? y : 0.f;
            y *= (float)maskO[ms];
            if (OUT_F32) {
                ((float*)outp)[(size_t)ms * COUT + cout] = y;   // un-haloed final output
            } else {
                int sod = ms / (OH * OW);
                int srm = ms - sod * (OH * OW);
                int soh = srm / OW;
                int sow = srm - soh * OW;
                size_t oidx = (((size_t)(sod + 1) * (OH + 2) + (soh + 1)) * (OW + 2)
                               + (sow + 1)) * COUT + cout;
                ((_Float16*)outp)[oidx] = (_Float16)y;
            }
        }
    }
    (void)ID;
}

// ---------------------------------------------------------------------------
// Host orchestration
// ---------------------------------------------------------------------------
static inline int cdiv_i(int a, int b) { return (a + b - 1) / b; }

extern "C" void kernel_launch(void* const* d_in, const int* in_sizes, int n_in,
                              void* d_out, int out_size, void* d_ws, size_t ws_size,
                              hipStream_t stream) {
    const float* feats  = (const float*)d_in[0];
    const int*   coords = (const int*)d_in[1];
    const float* W[9];
    const float* BN[9];
    for (int i = 0; i < 9; ++i) {
        W[i]  = (const float*)d_in[2 + 2 * i];
        BN[i] = (const float*)d_in[3 + 2 * i];
    }
    const int nvox = in_sizes[0] / 4;

    // Grid dims per level (logical, un-haloed) and haloed voxel counts
    const int D0 = 41, H0 = 264, W0 = 264;
    const int D2 = 20, H2 = 131, W2 = 131;
    const int D3 = 9,  H3 = 65,  W3 = 65;
    const int D4 = 4,  H4 = 32,  W4 = 32;
    const int D5 = 2,  H5 = 32,  W5 = 32;
    const int V0 = D0 * H0 * W0, V2 = D2 * H2 * W2, V3 = D3 * H3 * W3;
    const int V4 = D4 * H4 * W4, V5 = D5 * H5 * W5;
    const long long V0H = (long long)(D0 + 2) * (H0 + 2) * (W0 + 2);
    const long long V2H = (long long)(D2 + 2) * (H2 + 2) * (W2 + 2);
    const long long V3H = (long long)(D3 + 2) * (H3 + 2) * (W3 + 2);
    const long long V4H = (long long)(D4 + 2) * (H4 + 2) * (W4 + 2);

    // ---- workspace bump allocator (256B aligned) ----
    char* ws = (char*)d_ws;
    size_t off = 0;
    auto alloc = [&](size_t bytes) -> char* {
        char* p = ws + off;
        off += (bytes + 255) & ~(size_t)255;
        return p;
    };

    // Per-layer weight geometry: real cin, padded cin (matches kernel CIN), taps
    const int ntaps[9] = {27, 27, 27, 27, 27, 27, 27, 27, 3};
    const int cinR[9]  = {4, 16, 16, 32, 32, 64, 64, 64, 64};
    const int cinP[9]  = {16, 16, 16, 32, 32, 64, 64, 64, 64};
    const int coutN[9] = {16, 16, 32, 32, 64, 64, 64, 64, 128};
    int nchunk[9], wcntPad[9];
    for (int i = 0; i < 9; ++i) {
        nchunk[i]  = (ntaps[i] * cinP[i] + 31) / 32;
        wcntPad[i] = nchunk[i] * 32 * coutN[i];
    }

    _Float16* wh[9];
    float* sc[9];
    float* sh[9];
    for (int i = 0; i < 9; ++i) {
        wh[i] = (_Float16*)alloc((size_t)wcntPad[i] * 2);
        sc[i] = (float*)alloc((size_t)coutN[i] * 4);
        sh[i] = (float*)alloc((size_t)coutN[i] * 4);
    }
    _Float16* mask0 = (_Float16*)alloc((size_t)V0 * 2);
    _Float16* m2    = (_Float16*)alloc((size_t)V2 * 2);
    _Float16* m3    = (_Float16*)alloc((size_t)V3 * 2);
    _Float16* m4    = (_Float16*)alloc((size_t)V4 * 2);
    _Float16* mout  = (_Float16*)alloc((size_t)V5 * 2);
    const size_t featBytes = (size_t)V0H * 16 * 2;   // largest haloed activation (f16)
    _Float16* bufA = (_Float16*)alloc(featBytes);
    _Float16* bufB = (_Float16*)alloc(featBytes);
    (void)ws_size;

    auto fillw = [&](void* p, long long halves) {
        long long nw = halves / 2;
        fill_zero_w_kernel<<<(int)((nw + 255) / 256), 256, 0, stream>>>((unsigned*)p, nw);
    };

    // ---- 1. weight re-tiling (fragment-ready, cin+K zero-padded) + BN fold ----
    for (int i = 0; i < 9; ++i) {
        wtrans_kernel<<<cdiv_i(wcntPad[i], 256), 256, 0, stream>>>(
            W[i], wh[i], ntaps[i], cinR[i], cinP[i], coutN[i], nchunk[i]);
        bnprep_kernel<<<1, 128, 0, stream>>>(BN[i], sc[i], sh[i], coutN[i]);
    }

    // ---- 2. zero dense input grid (haloed, 16ch, in bufA) + mask0, scatter ----
    fillw(bufA, V0H * 16);
    fillw(mask0, V0);
    scatter_kernel<<<cdiv_i(nvox, 256), 256, 0, stream>>>(feats, coords, nvox, bufA, mask0,
                                                          H0, W0);

    // ---- 3. mask max-pool chain (un-haloed grids) ----
    maskpool_kernel<<<cdiv_i(V2, 256), 256, 0, stream>>>(mask0, H0, W0, m2, D2, H2, W2, 3, 3, 3, 2);
    maskpool_kernel<<<cdiv_i(V3, 256), 256, 0, stream>>>(m2, H2, W2, m3, D3, H3, W3, 3, 3, 3, 2);
    maskpool_kernel<<<cdiv_i(V4, 256), 256, 0, stream>>>(m3, H3, W3, m4, D4, H4, W4, 3, 3, 3, 2);
    maskpool_kernel<<<cdiv_i(V5, 256), 256, 0, stream>>>(m4, H4, W4, mout, D5, H5, W5, 3, 1, 1, 1);

    // ---- 4. conv chain (ping-pong; zero-fill output's haloed region first) ----
    // L1a: (4->pad16)->16 subm
    fillw(bufB, V0H * 16);
    conv3d_wmma<16, 16, 3, 3, 3, 1, 1, 1, 1, false>
        <<<dim3(cdiv_i(V0, 64), 1), 128, 0, stream>>>(bufA, D0, H0, W0, bufB, D0, H0, W0,
                                                      wh[0], sc[0], sh[0], mask0);
    // L1b: 16->16 subm
    fillw(bufA, V0H * 16);
    conv3d_wmma<16, 16, 3, 3, 3, 1, 1, 1, 1, false>
        <<<dim3(cdiv_i(V0, 64), 1), 128, 0, stream>>>(bufB, D0, H0, W0, bufA, D0, H0, W0,
                                                      wh[1], sc[1], sh[1], mask0);
    // L2a: 16->32 stride 2 VALID
    fillw(bufB, V2H * 32);
    conv3d_wmma<16, 32, 3, 3, 3, 2, 0, 0, 0, false>
        <<<dim3(cdiv_i(V2, 64), 2), 128, 0, stream>>>(bufA, D0, H0, W0, bufB, D2, H2, W2,
                                                      wh[2], sc[2], sh[2], m2);
    // L2b: 32->32 subm
    fillw(bufA, V2H * 32);
    conv3d_wmma<32, 32, 3, 3, 3, 1, 1, 1, 1, false>
        <<<dim3(cdiv_i(V2, 64), 2), 128, 0, stream>>>(bufB, D2, H2, W2, bufA, D2, H2, W2,
                                                      wh[3], sc[3], sh[3], m2);
    // L3a: 32->64 stride 2 VALID
    fillw(bufB, V3H * 64);
    conv3d_wmma<32, 64, 3, 3, 3, 2, 0, 0, 0, false>
        <<<dim3(cdiv_i(V3, 64), 4), 128, 0, stream>>>(bufA, D2, H2, W2, bufB, D3, H3, W3,
                                                      wh[4], sc[4], sh[4], m3);
    // L3b: 64->64 subm
    fillw(bufA, V3H * 64);
    conv3d_wmma<64, 64, 3, 3, 3, 1, 1, 1, 1, false>
        <<<dim3(cdiv_i(V3, 64), 4), 128, 0, stream>>>(bufB, D3, H3, W3, bufA, D3, H3, W3,
                                                      wh[5], sc[5], sh[5], m3);
    // L4a: 64->64 stride 2 VALID
    fillw(bufB, V4H * 64);
    conv3d_wmma<64, 64, 3, 3, 3, 2, 0, 0, 0, false>
        <<<dim3(cdiv_i(V4, 64), 4), 128, 0, stream>>>(bufA, D3, H3, W3, bufB, D4, H4, W4,
                                                      wh[6], sc[6], sh[6], m4);
    // L4b: 64->64 subm
    fillw(bufA, V4H * 64);
    conv3d_wmma<64, 64, 3, 3, 3, 1, 1, 1, 1, false>
        <<<dim3(cdiv_i(V4, 64), 4), 128, 0, stream>>>(bufB, D4, H4, W4, bufA, D4, H4, W4,
                                                      wh[7], sc[7], sh[7], m4);
    // Lout: 64->128, kernel (3,1,1), VALID, f32 straight to d_out (un-haloed)
    conv3d_wmma<64, 128, 3, 1, 1, 1, 0, 0, 0, true>
        <<<dim3(cdiv_i(V5, 64), 8), 128, 0, stream>>>(bufA, D4, H4, W4, d_out, D5, H5, W5,
                                                      wh[8], sc[8], sh[8], mout);
    (void)n_in;
    (void)out_size;
}
